// EncoderBlock_22290880266255
// MI455X (gfx1250) — matrix operand order repaired and checked by reading, not compiled, not run on hardware
//
#include <hip/hip_runtime.h>
#include <math.h>

// ---------------------------------------------------------------------------
// Types
// ---------------------------------------------------------------------------
typedef __bf16 bf16;
typedef __attribute__((ext_vector_type(16))) __bf16 bf16x16;
typedef __attribute__((ext_vector_type(8)))  __bf16 bf16x8;
typedef __attribute__((ext_vector_type(8)))  float  floatx8;
typedef __attribute__((ext_vector_type(4)))  float  floatx4;

#define D_MODEL 1024
#define HEADS   16
#define DKH     64
#define D_FF    4096
#define SEQ     2048
#define BATCH   4
#define ROWS    (BATCH*SEQ)   // 8192

#define BKSTEP  64            // K elements staged per LDS buffer
#define APAD    72            // padded LDS row length (bf16) -> bank-conflict-free frags

union BFrag { bf16x16 v16; bf16x8 v8[2]; };

__device__ __forceinline__ floatx8 wmma_bf16(const BFrag& a, const BFrag& b, floatx8 c) {
  // D = A(16x32) * B(32x16) + C ; fp32 accumulate
  return __builtin_amdgcn_wmma_f32_16x16x32_bf16(false, a.v16, false, b.v16,
                                                 (short)0, c, false, false);
}

// A fragment: lane L holds row (L%16); K elems at +8*(L/16) and +16+8*(L/16)
__device__ __forceinline__ void load_afrag(BFrag& f, const bf16* p) {
  f.v8[0] = *(const bf16x8*)(p);
  f.v8[1] = *(const bf16x8*)(p + 16);
}
// B fragment: lane L holds col (L%16); K elems at +8*(L/16) and +16+8*(L/16)
__device__ __forceinline__ void load_bfrag(BFrag& f, const bf16* p) {
  f.v8[0] = *(const bf16x8*)(p);
  f.v8[1] = *(const bf16x8*)(p + 16);
}

// Async DMA: copy 64 contiguous bytes global -> LDS (4 x b128), ASYNCcnt-tracked.
// INST_OFFSET is added to BOTH the LDS (VDST) and global (VADDR) addresses.
__device__ __forceinline__ void async_tile64(uint32_t lds, const bf16* g) {
  asm volatile(
      "global_load_async_to_lds_b128 %0, %1, off\n\t"
      "global_load_async_to_lds_b128 %0, %1, off offset:16\n\t"
      "global_load_async_to_lds_b128 %0, %1, off offset:32\n\t"
      "global_load_async_to_lds_b128 %0, %1, off offset:48"
      :: "v"(lds), "v"(g) : "memory");
}
__device__ __forceinline__ void wait_async0() {
  asm volatile("s_wait_asynccnt 0x0" ::: "memory");
}

// ---------------------------------------------------------------------------
// fp32 -> bf16 conversion (weights)
// ---------------------------------------------------------------------------
__global__ void cvt_f32_bf16_k(const float* __restrict__ in, bf16* __restrict__ out, int n) {
  int i = blockIdx.x * blockDim.x + threadIdx.x;
  int stride = gridDim.x * blockDim.x;
  for (; i < n; i += stride) out[i] = (bf16)in[i];
}

// ---------------------------------------------------------------------------
// LayerNorm over D_MODEL=1024, one row per 256-thread block, bf16 output
// ---------------------------------------------------------------------------
__global__ __launch_bounds__(256) void layernorm_bf16_k(
    const float* __restrict__ x, const float* __restrict__ alpha,
    const float* __restrict__ beta, bf16* __restrict__ out)
{
  __shared__ float red[8];
  __shared__ float stat[2];
  const int row = blockIdx.x, tid = threadIdx.x;
  const int lane = tid & 31, wave = tid >> 5;

  floatx4 v = *(const floatx4*)(x + (size_t)row * D_MODEL + tid * 4);
  float s = v[0] + v[1] + v[2] + v[3];
  for (int o = 16; o; o >>= 1) s += __shfl_xor(s, o, 32);
  if (lane == 0) red[wave] = s;
  __syncthreads();
  if (tid == 0) { float t = 0.f; for (int i = 0; i < 8; i++) t += red[i]; stat[0] = t * (1.0f / D_MODEL); }
  __syncthreads();
  const float mean = stat[0];

  float d[4]; float vs = 0.f;
  for (int i = 0; i < 4; i++) { d[i] = v[i] - mean; vs += d[i] * d[i]; }
  for (int o = 16; o; o >>= 1) vs += __shfl_xor(vs, o, 32);
  if (lane == 0) red[wave] = vs;
  __syncthreads();
  if (tid == 0) { float t = 0.f; for (int i = 0; i < 8; i++) t += red[i];
                  stat[1] = rsqrtf(t * (1.0f / D_MODEL) + 1e-6f); }
  __syncthreads();
  const float rstd = stat[1];

  const int c = tid * 4;
  bf16* o4 = out + (size_t)row * D_MODEL + c;
  for (int i = 0; i < 4; i++) o4[i] = (bf16)(alpha[c + i] * d[i] * rstd + beta[c + i]);
}

// ---------------------------------------------------------------------------
// bf16 WMMA GEMM: out[M,N] = A[M,K] @ Bw[N,K]^T (+bias)(ReLU)(+res)
// 256 thr = 8 waves; 128x128 block tile; wave: 64x32 (4x2 wmma tiles).
// K staged through double-buffered LDS via GLOBAL_LOAD_ASYNC_TO_LDS_B128,
// 64 K-elements per stage, one barrier per stage.
// ---------------------------------------------------------------------------
template<bool BIAS, bool RELU, bool RES, bool OUTBF>
__global__ __launch_bounds__(256) void gemm_bf16_k(
    const bf16* __restrict__ A, const bf16* __restrict__ Bw,
    const float* __restrict__ bias, const float* __restrict__ res,
    float* __restrict__ outf, bf16* __restrict__ outb,
    int N, int K)
{
  __shared__ __align__(16) bf16 aT[2][128][APAD];
  __shared__ __align__(16) bf16 bT[2][128][APAD];

  const int tid  = threadIdx.x;
  const int lane = tid & 31;
  const int wave = tid >> 5;
  const int wm   = wave >> 2;            // 0..1 -> 64-row slice
  const int wn   = wave & 3;             // 0..3 -> 32-col slice
  const int M0   = blockIdx.y * 128 + wm * 64;
  const int N0   = blockIdx.x * 128 + wn * 32;
  const int lr   = lane & 15;
  const int lh   = lane >> 4;

  // DMA assignment: thread t stages 32 elems (64B) of row (t>>1), half (t&1)
  const int ldr = tid >> 1;
  const int ldc = (tid & 1) * 32;
  const bf16* agp = A  + (size_t)(blockIdx.y * 128 + ldr) * K + ldc;
  const bf16* bgp = Bw + (size_t)(blockIdx.x * 128 + ldr) * K + ldc;
  uint32_t aLds[2], bLds[2];
  aLds[0] = (uint32_t)(uintptr_t)&aT[0][ldr][ldc];
  aLds[1] = (uint32_t)(uintptr_t)&aT[1][ldr][ldc];
  bLds[0] = (uint32_t)(uintptr_t)&bT[0][ldr][ldc];
  bLds[1] = (uint32_t)(uintptr_t)&bT[1][ldr][ldc];

  floatx8 acc[4][2];
  for (int i = 0; i < 4; i++) for (int j = 0; j < 2; j++) acc[i][j] = (floatx8)0.0f;

  // prologue: stage K[0:64) into buffer 0
  async_tile64(aLds[0], agp);
  async_tile64(bLds[0], bgp);
  wait_async0();
  __syncthreads();

  int buf = 0;
  for (int k0 = 0; k0 < K; k0 += BKSTEP) {
    // prefetch next stage into the other buffer while we compute
    if (k0 + BKSTEP < K) {
      async_tile64(aLds[buf ^ 1], agp + k0 + BKSTEP);
      async_tile64(bLds[buf ^ 1], bgp + k0 + BKSTEP);
    }
    // consume current buffer: two 32-K chunks
    #pragma unroll
    for (int c = 0; c < 2; c++) {
      BFrag af[4], bfr[2];
      for (int i = 0; i < 4; i++)
        load_afrag(af[i], &aT[buf][wm * 64 + i * 16 + lr][c * 32 + 8 * lh]);
      for (int j = 0; j < 2; j++)
        load_bfrag(bfr[j], &bT[buf][wn * 32 + j * 16 + lr][c * 32 + 8 * lh]);
      for (int i = 0; i < 4; i++)
        for (int j = 0; j < 2; j++)
          acc[i][j] = wmma_bf16(af[i], bfr[j], acc[i][j]);
    }
    wait_async0();       // our DMA into buf^1 landed
    __syncthreads();     // everyone done reading buf & done DMAing
    buf ^= 1;
  }

  // epilogue: C layout -> lane holds (M = mt + v + 8*lh, N = nt + lr) for v=0..7
  for (int j = 0; j < 2; j++) {
    const int col = N0 + j * 16 + lr;
    float bv = 0.0f;
    if (BIAS) bv = bias[col];
    for (int i = 0; i < 4; i++) {
      const int rbase = M0 + i * 16 + 8 * lh;
      for (int v = 0; v < 8; v++) {
        const size_t idx = (size_t)(rbase + v) * N + col;
        float val = acc[i][j][v];
        if (BIAS) val += bv;
        if (RELU) val = fmaxf(val, 0.0f);
        if (RES)  val += res[idx];
        if (OUTBF) outb[idx] = (bf16)val;
        else       outf[idx] = val;
      }
    }
  }
}

// ---------------------------------------------------------------------------
// V transpose: v[b*S+s, h*64+d] -> vT[((b*H+h)*64+d)*S + s]
// ---------------------------------------------------------------------------
__global__ void transpose_v_k(const bf16* __restrict__ v, bf16* __restrict__ vT) {
  int idx = blockIdx.x * blockDim.x + threadIdx.x;
  if (idx >= ROWS * D_MODEL) return;
  const int s  = idx & (SEQ - 1);
  const int d  = (idx >> 11) & (DKH - 1);
  const int bh = idx >> 17;
  const int b = bh >> 4, h = bh & 15;
  vT[idx] = v[((size_t)(b * SEQ + s)) * D_MODEL + h * DKH + d];
}

// ---------------------------------------------------------------------------
// Flash attention, one wave per 16-row Q tile, 32 keys / iteration.
// block = 128 thr (4 waves); grid = (S/64, B*H)
// ---------------------------------------------------------------------------
__global__ __launch_bounds__(128) void flash_attn_k(
    const bf16* __restrict__ q, const bf16* __restrict__ k,
    const bf16* __restrict__ vT, const int* __restrict__ mask,
    bf16* __restrict__ ctx)
{
  const int tid  = threadIdx.x;
  const int lane = tid & 31, wave = tid >> 5;
  const int lr   = lane & 15, lh = lane >> 4;
  const int bh   = blockIdx.y;
  const int b    = bh >> 4, h = bh & 15;
  const int q0   = blockIdx.x * 64 + wave * 16;
  const size_t r0 = (size_t)b * SEQ + q0;

  __shared__ __align__(16) bf16 pbuf[4][16][32];

  // Q fragments: 16x64 as two 16x32 A-frags
  BFrag qa[2];
  for (int dk = 0; dk < 2; dk++)
    load_afrag(qa[dk], q + (r0 + lr) * D_MODEL + h * DKH + dk * 32 + 8 * lh);

  float m[8], l[8], sc[8];
  floatx8 o[4];
  for (int v = 0; v < 8; v++) { m[v] = -3e38f; l[v] = 0.f; }
  for (int t = 0; t < 4; t++) o[t] = (floatx8)0.0f;

  const int* mrow = mask + b * SEQ;

  for (int kc = 0; kc < SEQ / 32; kc++) {
    const int kbase = kc * 32;

    // scores S = Q @ K^T : B operand from k (row-major [key, d], d-contiguous)
    BFrag kf[2][2];
    for (int j = 0; j < 2; j++) {
      const bf16* kp = k + ((size_t)b * SEQ + kbase + j * 16 + lr) * D_MODEL + h * DKH + 8 * lh;
      load_bfrag(kf[j][0], kp);
      load_bfrag(kf[j][1], kp + 32);
    }
    floatx8 s0 = (floatx8)0.0f, s1 = (floatx8)0.0f;
    s0 = wmma_bf16(qa[0], kf[0][0], s0);
    s0 = wmma_bf16(qa[1], kf[0][1], s0);
    s1 = wmma_bf16(qa[0], kf[1][0], s1);
    s1 = wmma_bf16(qa[1], kf[1][1], s1);

    // scale 1/sqrt(64) + mask
    const int mk0 = mrow[kbase + lr];
    const int mk1 = mrow[kbase + 16 + lr];
    for (int v = 0; v < 8; v++) {
      s0[v] = (mk0 == 0) ? -3e38f : s0[v] * 0.125f;
      s1[v] = (mk1 == 0) ? -3e38f : s1[v] * 0.125f;
    }

    // online softmax; row M = v + 8*lh spans 16 lanes of a half-wave
    for (int v = 0; v < 8; v++) {
      float mv = fmaxf(s0[v], s1[v]);
      mv = fmaxf(mv, __shfl_xor(mv, 1, 32));
      mv = fmaxf(mv, __shfl_xor(mv, 2, 32));
      mv = fmaxf(mv, __shfl_xor(mv, 4, 32));
      mv = fmaxf(mv, __shfl_xor(mv, 8, 32));
      const float mn = fmaxf(m[v], mv);
      sc[v] = __expf(m[v] - mn);
      m[v]  = mn;
      const float p0 = __expf(s0[v] - mn);
      const float p1 = __expf(s1[v] - mn);
      s0[v] = p0; s1[v] = p1;
      float rs = p0 + p1;
      rs += __shfl_xor(rs, 1, 32);
      rs += __shfl_xor(rs, 2, 32);
      rs += __shfl_xor(rs, 4, 32);
      rs += __shfl_xor(rs, 8, 32);
      l[v] = l[v] * sc[v] + rs;
    }
    for (int t = 0; t < 4; t++)
      for (int v = 0; v < 8; v++) o[t][v] *= sc[v];

    // P (C layout) -> LDS row-major [16 q rows][32 keys] -> reload as A-frag
    for (int v = 0; v < 8; v++) {
      pbuf[wave][v + 8 * lh][lr]      = (bf16)s0[v];
      pbuf[wave][v + 8 * lh][16 + lr] = (bf16)s1[v];
    }
    BFrag pa;
    load_afrag(pa, &pbuf[wave][lr][8 * lh]);

    // V fragments from vT (key-contiguous rows)
    BFrag vf[4];
    for (int t = 0; t < 4; t++)
      load_bfrag(vf[t], vT + ((size_t)bh * DKH + t * 16 + lr) * SEQ + kbase + 8 * lh);
    for (int t = 0; t < 4; t++) o[t] = wmma_bf16(pa, vf[t], o[t]);
  }

  for (int v = 0; v < 8; v++) l[v] = (l[v] > 0.f) ? 1.0f / l[v] : 0.f;
  for (int t = 0; t < 4; t++)
    for (int v = 0; v < 8; v++)
      ctx[(r0 + v + 8 * lh) * D_MODEL + h * DKH + t * 16 + lr] = (bf16)(o[t][v] * l[v]);
}

// ---------------------------------------------------------------------------
// Launch
// ---------------------------------------------------------------------------
extern "C" void kernel_launch(void* const* d_in, const int* in_sizes, int n_in,
                              void* d_out, int out_size, void* d_ws, size_t ws_size,
                              hipStream_t stream)
{
  (void)in_sizes; (void)n_in; (void)out_size; (void)ws_size;
  const float* x    = (const float*)d_in[0];
  const int*   mask = (const int*)  d_in[1];
  const float* wq   = (const float*)d_in[2];
  const float* wk   = (const float*)d_in[3];
  const float* wv   = (const float*)d_in[4];
  const float* wo   = (const float*)d_in[5];
  const float* ln1a = (const float*)d_in[6];
  const float* ln1b = (const float*)d_in[7];
  const float* ln2a = (const float*)d_in[8];
  const float* ln2b = (const float*)d_in[9];
  const float* w1   = (const float*)d_in[10];
  const float* b1   = (const float*)d_in[11];
  const float* w2   = (const float*)d_in[12];
  const float* b2   = (const float*)d_in[13];
  float* out = (float*)d_out;

  char* ws = (char*)d_ws;
  size_t off = 0;
  auto alloc = [&](size_t bytes) -> void* {
    void* p = ws + off;
    off = (off + bytes + 255) & ~(size_t)255;
    return p;
  };
  bf16* wq_b = (bf16*)alloc((size_t)D_MODEL * D_MODEL * 2);
  bf16* wk_b = (bf16*)alloc((size_t)D_MODEL * D_MODEL * 2);
  bf16* wv_b = (bf16*)alloc((size_t)D_MODEL * D_MODEL * 2);
  bf16* wo_b = (bf16*)alloc((size_t)D_MODEL * D_MODEL * 2);
  bf16* w1_b = (bf16*)alloc((size_t)D_FF * D_MODEL * 2);
  bf16* w2_b = (bf16*)alloc((size_t)D_MODEL * D_FF * 2);
  bf16* xn_b = (bf16*)alloc((size_t)ROWS * D_MODEL * 2);
  bf16* q_b  = (bf16*)alloc((size_t)ROWS * D_MODEL * 2);
  bf16* k_b  = (bf16*)alloc((size_t)ROWS * D_MODEL * 2);
  bf16* v_b  = (bf16*)alloc((size_t)ROWS * D_MODEL * 2);
  bf16* vT_b = (bf16*)alloc((size_t)ROWS * D_MODEL * 2);
  bf16* ctx_b= (bf16*)alloc((size_t)ROWS * D_MODEL * 2);
  float* x1  = (float*)alloc((size_t)ROWS * D_MODEL * 4);
  bf16* h_b  = (bf16*)alloc((size_t)ROWS * D_FF * 2);

  // 1) weights -> bf16
  cvt_f32_bf16_k<<<2048, 256, 0, stream>>>(wq, wq_b, D_MODEL * D_MODEL);
  cvt_f32_bf16_k<<<2048, 256, 0, stream>>>(wk, wk_b, D_MODEL * D_MODEL);
  cvt_f32_bf16_k<<<2048, 256, 0, stream>>>(wv, wv_b, D_MODEL * D_MODEL);
  cvt_f32_bf16_k<<<2048, 256, 0, stream>>>(wo, wo_b, D_MODEL * D_MODEL);
  cvt_f32_bf16_k<<<4096, 256, 0, stream>>>(w1, w1_b, D_FF * D_MODEL);
  cvt_f32_bf16_k<<<4096, 256, 0, stream>>>(w2, w2_b, D_MODEL * D_FF);

  // 2) LN1
  layernorm_bf16_k<<<ROWS, 256, 0, stream>>>(x, ln1a, ln1b, xn_b);

  // 3) Q, K, V projections (M=8192, N=1024, K=1024)
  dim3 g1(D_MODEL / 128, ROWS / 128);
  gemm_bf16_k<false,false,false,true><<<g1, 256, 0, stream>>>(xn_b, wq_b, nullptr, nullptr, nullptr, q_b, D_MODEL, D_MODEL);
  gemm_bf16_k<false,false,false,true><<<g1, 256, 0, stream>>>(xn_b, wk_b, nullptr, nullptr, nullptr, k_b, D_MODEL, D_MODEL);
  gemm_bf16_k<false,false,false,true><<<g1, 256, 0, stream>>>(xn_b, wv_b, nullptr, nullptr, nullptr, v_b, D_MODEL, D_MODEL);

  // 4) V -> per-head transposed layout
  transpose_v_k<<<(ROWS * D_MODEL + 255) / 256, 256, 0, stream>>>(v_b, vT_b);

  // 5) flash attention
  flash_attn_k<<<dim3(SEQ / 64, BATCH * HEADS), 128, 0, stream>>>(q_b, k_b, vT_b, mask, ctx_b);

  // 6) output projection + residual (fp32 out)
  gemm_bf16_k<false,false,true,false><<<g1, 256, 0, stream>>>(ctx_b, wo_b, nullptr, x, x1, nullptr, D_MODEL, D_MODEL);

  // 7) LN2
  layernorm_bf16_k<<<ROWS, 256, 0, stream>>>(x1, ln2a, ln2b, xn_b);

  // 8) FFN1: relu(xn @ w1.T + b1) -> bf16 (M=8192, N=4096, K=1024)
  dim3 g2(D_FF / 128, ROWS / 128);
  gemm_bf16_k<true,true,false,true><<<g2, 256, 0, stream>>>(xn_b, w1_b, b1, nullptr, nullptr, h_b, D_FF, D_MODEL);

  // 9) FFN2: x1 + (h @ w2.T + b2) -> fp32 out (M=8192, N=1024, K=4096)
  gemm_bf16_k<true,false,true,false><<<g1, 256, 0, stream>>>(h_b, w2_b, b2, x1, out, nullptr, D_MODEL, D_FF);
}
